// MultiHeadLatentAttention_47236050321675
// MI455X (gfx1250) — compile-verified, matrix operand order checked by reading, not run on hardware
//
#include <hip/hip_runtime.h>
#include <cmath>

typedef __bf16 bf16;
typedef __attribute__((ext_vector_type(16))) __bf16 v16bf;
typedef __attribute__((ext_vector_type(8)))  float  v8f;
typedef __attribute__((ext_vector_type(4)))  unsigned int u32x4;
typedef __attribute__((ext_vector_type(8)))  int i32x8;
typedef __attribute__((ext_vector_type(4)))  int i32x4;

// K index inside a 16x32 bf16 A-fragment (and symmetric B-fragment):
//   lanes 0-15 : VGPR0-3 -> K 0..7,  VGPR4-7 -> K 16..23
//   lanes16-31 : VGPR0-3 -> K 8..15, VGPR4-7 -> K 24..31
__device__ __forceinline__ int wmma_k(int i, int half) {
  int v = i >> 1;
  return ((v < 4) ? 0 : 16) + (half ? 8 : 0) + 2 * (v & 3) + (i & 1);
}

// ---------------------------------------------------------------------------
// TDM: issue a 2D tile load (bf16 elements) global -> LDS via a D# descriptor.
// width_elems = tile_dim0 (row length), rows = tile_dim1,
// row_stride_elems = tensor_dim0_stride. LDS receives rows contiguously.
// ---------------------------------------------------------------------------
__device__ __forceinline__ void tdm_load_2d_bf16(unsigned lds_byte_addr,
                                                 const void* gptr,
                                                 int width_elems, int rows,
                                                 int row_stride_elems)
{
  unsigned long long ga = (unsigned long long)(size_t)gptr;
  u32x4 g0;
  g0[0] = 1u;                                    // count=1, user descriptor
  g0[1] = lds_byte_addr;                         // lds_addr
  g0[2] = (unsigned)(ga & 0xFFFFFFFFull);        // global_addr[31:0]
  g0[3] = ((unsigned)(ga >> 32) & 0x01FFFFFFu) | (2u << 30);  // addr[56:32] | type=2
  i32x8 g1;
  g1[0] = (1 << 16);                                           // data_size = 2B
  g1[1] = (width_elems & 0xFFFF) << 16;                        // tensor_dim0 lo
  g1[2] = ((unsigned)width_elems >> 16) | ((rows & 0xFFFF) << 16);   // dim0 hi | tensor_dim1 lo
  g1[3] = ((unsigned)rows >> 16) | ((width_elems & 0xFFFF) << 16);   // dim1 hi | tile_dim0
  g1[4] = rows & 0xFFFF;                                       // tile_dim1 | tile_dim2=0
  g1[5] = row_stride_elems;                                    // tensor_dim0_stride lo
  g1[6] = 0;                                                   // stride hi | dim1_stride lo
  g1[7] = 0;
  i32x4 z4 = {0, 0, 0, 0};
#if defined(__clang_major__) && (__clang_major__ >= 23)
  i32x8 z8 = {0, 0, 0, 0, 0, 0, 0, 0};
  __builtin_amdgcn_tensor_load_to_lds(g0, g1, z4, z4, z8, 0);
#else
  __builtin_amdgcn_tensor_load_to_lds(g0, g1, z4, z4, 0);
#endif
}

// ---------------------------------------------------------------------------
// Generic GEMM: Out[m, remap(n)] = sum_k A[m,k] * W[n,k] + bias[n]
// Block tile 64x128, K-step 32, 8 waves (2Mx4N), each wave 2x2 16x16 frags.
// Store remap: h = n / head_in; d = n % head_in; col = h*head_stride+head_base+d
// ---------------------------------------------------------------------------
#define GTM 64
#define GTN 128
#define GTK 32
#define ASTR (GTK + 4)
#define BSTR (GTK + 4)

template <typename AT, typename OutT>
__global__ __launch_bounds__(256)
void gemm_bf16_kernel(const AT* __restrict__ A, const float* __restrict__ W,
                      const float* __restrict__ bias, OutT* __restrict__ Out,
                      int M, int N, int K,
                      int head_in, int head_stride, int head_base, int out_row_stride)
{
  __shared__ bf16 As[GTM * ASTR];
  __shared__ bf16 Bs[GTN * BSTR];

  const int tid  = threadIdx.x;
  const int wave = tid >> 5;
  const int lane = tid & 31;
  const int half = lane >> 4;
  const int lrow = lane & 15;

  const int m0 = blockIdx.y * GTM;
  const int n0 = blockIdx.x * GTN;
  const int waveM = (wave >> 2) * 32;
  const int waveN = (wave & 3) * 32;

  int   cofs[2];
  float binit[2];
#pragma unroll
  for (int nf = 0; nf < 2; ++nf) {
    int n_g = n0 + waveN + nf * 16 + lrow;
    int h = n_g / head_in;
    int d = n_g - h * head_in;
    cofs[nf]  = h * head_stride + head_base + d;
    binit[nf] = bias[n_g];
  }

  v8f acc[2][2];
#pragma unroll
  for (int mf = 0; mf < 2; ++mf)
#pragma unroll
    for (int nf = 0; nf < 2; ++nf)
#pragma unroll
      for (int i = 0; i < 8; ++i)
        acc[mf][nf][i] = binit[nf];

  for (int k0 = 0; k0 < K; k0 += GTK) {
    __syncthreads();
    for (int idx = tid; idx < GTM * GTK; idx += 256) {
      int r = idx >> 5, c = idx & 31;
      As[r * ASTR + c] = (bf16)A[(size_t)(m0 + r) * K + k0 + c];
    }
    for (int idx = tid; idx < GTN * GTK; idx += 256) {
      int r = idx >> 5, c = idx & 31;
      Bs[r * BSTR + c] = (bf16)W[(size_t)(n0 + r) * K + k0 + c];
    }
    if (k0 + GTK < K) {   // prefetch next K-slab -> global_prefetch_b8
      int r = tid >> 2;
      __builtin_prefetch(&A[(size_t)(m0 + r) * K + k0 + GTK + (tid & 3) * 8], 0, 1);
      __builtin_prefetch(&W[(size_t)(n0 + (tid >> 1)) * K + k0 + GTK + (tid & 1) * 16], 0, 1);
    }
    __syncthreads();

    v16bf afrag[2], bfrag[2];
#pragma unroll
    for (int mf = 0; mf < 2; ++mf) {
      int row = waveM + mf * 16 + lrow;
#pragma unroll
      for (int i = 0; i < 16; ++i)
        afrag[mf][i] = As[row * ASTR + wmma_k(i, half)];
    }
#pragma unroll
    for (int nf = 0; nf < 2; ++nf) {
      int col = waveN + nf * 16 + lrow;
#pragma unroll
      for (int i = 0; i < 16; ++i)
        bfrag[nf][i] = Bs[col * BSTR + wmma_k(i, half)];
    }
#pragma unroll
    for (int mf = 0; mf < 2; ++mf)
#pragma unroll
      for (int nf = 0; nf < 2; ++nf)
        acc[mf][nf] = __builtin_amdgcn_wmma_f32_16x16x32_bf16(
            false, afrag[mf], false, bfrag[nf], (short)0, acc[mf][nf], false, false);
  }

#pragma unroll
  for (int mf = 0; mf < 2; ++mf)
#pragma unroll
    for (int nf = 0; nf < 2; ++nf)
#pragma unroll
      for (int r = 0; r < 8; ++r) {
        int m_g = m0 + waveM + mf * 16 + r + half * 8;
        Out[(size_t)m_g * out_row_stride + cofs[nf]] = (OutT)acc[mf][nf][r];
      }
}

// ---------------------------------------------------------------------------
// In-place RoPE on the [base, base+dr) slice of every (row, head), bf16 buffer
// ---------------------------------------------------------------------------
__global__ void rope_bf16_kernel(bf16* __restrict__ p, int S_, int rows, int heads,
                                 int row_stride, int head_stride, int base, int dr)
{
  int idx = blockIdx.x * blockDim.x + threadIdx.x;
  int half_d = dr >> 1;
  int total = rows * heads * half_d;
  if (idx >= total) return;
  int i   = idx % half_d;
  int t   = idx / half_d;
  int h   = t % heads;
  int row = t / heads;
  int s   = row % S_;
  float inv_freq = __powf(10000.f, -(2.f * (float)i) / (float)dr);
  float ang = (float)s * inv_freq;
  float sn, cs;
  __sincosf(ang, &sn, &cs);
  bf16* q = p + (size_t)row * row_stride + h * head_stride + base + 2 * i;
  float x1 = (float)q[0], x2 = (float)q[1];
  q[0] = (bf16)(x1 * cs - x2 * sn);
  q[1] = (bf16)(x1 * sn + x2 * cs);
}

// ---------------------------------------------------------------------------
// Flash attention, d=192 (QK) / 128 (V). 4 waves/WG, 16 q-rows per wave.
// K/V tiles (bf16) streamed by the Tensor Data Mover, double-buffered.
// ---------------------------------------------------------------------------
#define BC   32
#define DQK  192
#define DV   128

__global__ __launch_bounds__(128)
void mla_attn_kernel(const bf16* __restrict__ Q, const bf16* __restrict__ Kc,
                     const bf16* __restrict__ V, bf16* __restrict__ O,
                     int S_, float scale)
{
  __shared__ bf16 Ks[2][BC * DQK];   // 2 x 12 KB, TDM-filled (rows contiguous)
  __shared__ bf16 Vs[2][BC * DV];    // 2 x  8 KB
  __shared__ bf16 Ps[4 * 16 * 34];   // per-wave P staging

  const int tid  = threadIdx.x;
  const int wave = tid >> 5;
  const int lane = tid & 31;
  const int half = lane >> 4;
  const int lrow = lane & 15;

  const int b  = blockIdx.z;
  const int h  = blockIdx.y;
  const int q0 = blockIdx.x * 64 + wave * 16;

  const int qk_row_stride = 16 * DQK;   // H * 192
  const int v_row_stride  = 16 * DV;    // H * 128

  // Preload Q fragments: 16 rows x 192, six 16x32 A-frags, resident in VGPRs
  v16bf qf[6];
  {
    const bf16* qp = Q + ((size_t)b * S_ + q0 + lrow) * qk_row_stride + h * DQK;
#pragma unroll
    for (int ks = 0; ks < 6; ++ks)
#pragma unroll
      for (int i = 0; i < 16; ++i)
        qf[ks][i] = qp[ks * 32 + wmma_k(i, half)];
  }

  v8f ctx[8];
#pragma unroll
  for (int f = 0; f < 8; ++f)
#pragma unroll
    for (int i = 0; i < 8; ++i) ctx[f][i] = 0.f;
  float mrow[8], lsum[8];
#pragma unroll
  for (int r = 0; r < 8; ++r) { mrow[r] = -3.0e38f; lsum[r] = 0.f; }

  bf16* myPs = Ps + wave * 16 * 34;
  const int nblocks = S_ / BC;

  // TDM issue of key-block jb into buffer buf (wave 0 only)
  auto issue_tile = [&](int jb, int buf) {
    const bf16* kp = Kc + ((size_t)b * S_ + jb * BC) * qk_row_stride + h * DQK;
    const bf16* vp = V  + ((size_t)b * S_ + jb * BC) * v_row_stride  + h * DV;
    tdm_load_2d_bf16((unsigned)(size_t)&Ks[buf][0], kp, DQK, BC, qk_row_stride);
    tdm_load_2d_bf16((unsigned)(size_t)&Vs[buf][0], vp, DV,  BC, v_row_stride);
  };

  if (wave == 0) issue_tile(0, 0);

  for (int jb = 0; jb < nblocks; ++jb) {
    __syncthreads();   // prev iteration done reading the buffer we will refill
    if (wave == 0) {
      if (jb + 1 < nblocks) {
        issue_tile(jb + 1, (jb + 1) & 1);
        __builtin_amdgcn_s_wait_tensorcnt((short)2);   // tile jb complete
      } else {
        __builtin_amdgcn_s_wait_tensorcnt((short)0);
      }
    }
    __syncthreads();   // tile jb visible to all waves
    const bf16* Kt = &Ks[jb & 1][0];
    const bf16* Vt = &Vs[jb & 1][0];

    // S = Q (16x192) x K^T (192x32)
    v8f sf[2];
#pragma unroll
    for (int nf = 0; nf < 2; ++nf)
#pragma unroll
      for (int i = 0; i < 8; ++i) sf[nf][i] = 0.f;
#pragma unroll
    for (int ks = 0; ks < 6; ++ks)
#pragma unroll
      for (int nf = 0; nf < 2; ++nf) {
        v16bf kf;
        int key = nf * 16 + lrow;
#pragma unroll
        for (int i = 0; i < 16; ++i)
          kf[i] = Kt[key * DQK + ks * 32 + wmma_k(i, half)];
        sf[nf] = __builtin_amdgcn_wmma_f32_16x16x32_bf16(
            false, qf[ks], false, kf, (short)0, sf[nf], false, false);
      }

    // Online softmax; rows live across 16-lane groups (fully unrolled so all
    // vector-element accesses are static VGPRs -- no M0/movrels indexing)
#pragma unroll
    for (int r = 0; r < 8; ++r) {
      float s0 = sf[0][r] * scale;
      float s1 = sf[1][r] * scale;
      float rv = fmaxf(s0, s1);
#pragma unroll
      for (int off = 8; off >= 1; off >>= 1)
        rv = fmaxf(rv, __shfl_xor(rv, off, 32));
      float mnew  = fmaxf(mrow[r], rv);
      float alpha = __expf(mrow[r] - mnew);
      float p0 = __expf(s0 - mnew);
      float p1 = __expf(s1 - mnew);
      float rs = p0 + p1;
#pragma unroll
      for (int off = 8; off >= 1; off >>= 1)
        rs += __shfl_xor(rs, off, 32);
      lsum[r] = lsum[r] * alpha + rs;
      mrow[r] = mnew;
#pragma unroll
      for (int f = 0; f < 8; ++f) ctx[f][r] *= alpha;
      int prow = r + half * 8;
      myPs[prow * 34 + lrow]      = (bf16)p0;
      myPs[prow * 34 + 16 + lrow] = (bf16)p1;
    }
    __syncthreads();   // P staged

    // ctx += P (16x32) x V (32x128)
    v16bf pa;
#pragma unroll
    for (int i = 0; i < 16; ++i)
      pa[i] = myPs[lrow * 34 + wmma_k(i, half)];
#pragma unroll
    for (int f = 0; f < 8; ++f) {
      v16bf vf;
#pragma unroll
      for (int i = 0; i < 16; ++i)
        vf[i] = Vt[wmma_k(i, half) * DV + f * 16 + lrow];
      ctx[f] = __builtin_amdgcn_wmma_f32_16x16x32_bf16(
          false, pa, false, vf, (short)0, ctx[f], false, false);
    }
  }

#pragma unroll
  for (int r = 0; r < 8; ++r) {
    float inv = 1.f / lsum[r];
    int m_g = q0 + r + half * 8;
    size_t orow = ((size_t)b * S_ + m_g) * v_row_stride + h * DV;
#pragma unroll
    for (int f = 0; f < 8; ++f)
      O[orow + f * 16 + lrow] = (bf16)(ctx[f][r] * inv);
  }
}

// ---------------------------------------------------------------------------
extern "C" void kernel_launch(void* const* d_in, const int* in_sizes, int n_in,
                              void* d_out, int out_size, void* d_ws, size_t ws_size,
                              hipStream_t stream)
{
  const float* x      = (const float*)d_in[0];
  const float* d_kv_w = (const float*)d_in[1];
  const float* d_kv_b = (const float*)d_in[2];
  const float* u_k_w  = (const float*)d_in[3];
  const float* u_k_b  = (const float*)d_in[4];
  const float* u_v_w  = (const float*)d_in[5];
  const float* u_v_b  = (const float*)d_in[6];
  const float* d_q_w  = (const float*)d_in[7];
  const float* d_q_b  = (const float*)d_in[8];
  const float* u_q_w  = (const float*)d_in[9];
  const float* u_q_b  = (const float*)d_in[10];
  const float* qr_w   = (const float*)d_in[11];
  const float* qr_b   = (const float*)d_in[12];
  const float* out_w  = (const float*)d_in[13];
  const float* out_b  = (const float*)d_in[14];
  float* out = (float*)d_out;

  const int Bb = 2, Sb = 2048, Hh = 16, Dh = 128, Dr = 64, HID = 2048, Cc = 512;
  const int M = Bb * Sb;          // 4096
  const int Dcat = Dh + Dr;       // 192

  // All intermediates in bf16 (halves traffic; k_cat+v ~42MB -> L2 resident)
  bf16* kv_c  = (bf16*)d_ws;                          // M*C
  bf16* q_lat = kv_c  + (size_t)M * Cc;               // M*C
  bf16* q_cat = q_lat + (size_t)M * Cc;               // M*H*192
  bf16* k_cat = q_cat + (size_t)M * Hh * Dcat;        // M*H*192
  bf16* v_buf = k_cat + (size_t)M * Hh * Dcat;        // M*H*128
  bf16* ctx   = v_buf + (size_t)M * Hh * Dh;          // M*H*128

  dim3 blk(256);
  dim3 gC (Cc  / GTN, M / GTM);
  dim3 gKD(Hh * Dh / GTN, M / GTM);
  dim3 gKR(Hh * Dr / GTN, M / GTM);
  dim3 gH (HID / GTN, M / GTM);

  // down projections (fp32 activations -> bf16 latents)
  gemm_bf16_kernel<float, bf16><<<gC, blk, 0, stream>>>(
      x, d_kv_w, d_kv_b, kv_c, M, Cc, HID, Cc, Cc, 0, Cc);
  gemm_bf16_kernel<float, bf16><<<gC, blk, 0, stream>>>(
      x, d_q_w, d_q_b, q_lat, M, Cc, HID, Cc, Cc, 0, Cc);
  // up projections, remapped straight into concat buffers
  gemm_bf16_kernel<bf16, bf16><<<gKD, blk, 0, stream>>>(
      kv_c, u_k_w, u_k_b, k_cat, M, Hh * Dh, Cc, Dh, Dcat, 0, Hh * Dcat);
  gemm_bf16_kernel<bf16, bf16><<<gKR, blk, 0, stream>>>(
      kv_c, qr_w, qr_b, k_cat, M, Hh * Dr, Cc, Dr, Dcat, Dh, Hh * Dcat);
  gemm_bf16_kernel<bf16, bf16><<<gKD, blk, 0, stream>>>(
      kv_c, u_v_w, u_v_b, v_buf, M, Hh * Dh, Cc, Hh * Dh, Hh * Dh, 0, Hh * Dh);
  gemm_bf16_kernel<bf16, bf16><<<gKD, blk, 0, stream>>>(
      q_lat, u_q_w, u_q_b, q_cat, M, Hh * Dh, Cc, Dh, Dcat, 0, Hh * Dcat);
  gemm_bf16_kernel<bf16, bf16><<<gKR, blk, 0, stream>>>(
      q_lat, qr_w, qr_b, q_cat, M, Hh * Dr, Cc, Dr, Dcat, Dh, Hh * Dcat);

  // RoPE on the rotary slices
  int rope_total = M * Hh * (Dr / 2);
  int rgrid = (rope_total + 255) / 256;
  rope_bf16_kernel<<<rgrid, 256, 0, stream>>>(q_cat, Sb, M, Hh, Hh * Dcat, Dcat, Dh, Dr);
  rope_bf16_kernel<<<rgrid, 256, 0, stream>>>(k_cat, Sb, M, Hh, Hh * Dcat, Dcat, Dh, Dr);

  // attention (TDM-streamed K/V tiles)
  float scale = 1.0f / sqrtf((float)Dcat);
  dim3 agrid(Sb / 64, Hh, Bb);
  mla_attn_kernel<<<agrid, dim3(128), 0, stream>>>(q_cat, k_cat, v_buf, ctx, Sb, scale);

  // output projection (bf16 ctx -> fp32 out)
  gemm_bf16_kernel<bf16, float><<<gH, blk, 0, stream>>>(
      ctx, out_w, out_b, out, M, HID, Hh * Dh, HID, HID, 0, HID);
}
